// QBWTransition_28381143892471
// MI455X (gfx1250) — matrix operand order repaired and checked by reading, not compiled, not run on hardware
//
#include <hip/hip_runtime.h>
#include <stdint.h>

#define B_   32
#define CIN  512
#define HH   56
#define WW   56
#define HW   3136
#define COUT 256
#define PH   28
#define PQ   784
#define KSTEPS (CIN / 32)
#define EPSF 1e-5f

#define HSZ (64 * 20)    // uints per h-tile buffer  (64 px x (32 f16 + pad))
#define WSZ (32 * 132)   // uints per w-tile buffer  (32 K x (256 f16 + pad))

typedef __attribute__((ext_vector_type(16))) _Float16 v16h;
typedef __attribute__((ext_vector_type(8)))  float    v8f;
typedef __attribute__((ext_vector_type(4)))  int      v4i;

union Frag32B { unsigned int u[8]; v16h v; };

#if defined(__HIP_DEVICE_COMPILE__) &&                                        \
    __has_builtin(__builtin_amdgcn_global_load_async_to_lds_b128) &&          \
    __has_builtin(__builtin_amdgcn_s_wait_asynccnt)
#define USE_ASYNC_LDS 1
#define AS1 __attribute__((address_space(1)))
#define AS3 __attribute__((address_space(3)))
// src: global v4i* (as1), dst: LDS v4i* (as3), imm offset, imm cpol
__device__ __forceinline__ void async_b128(const void* gsrc, void* ldst) {
  __builtin_amdgcn_global_load_async_to_lds_b128(
      (AS1 v4i*)(v4i*)gsrc, (AS3 v4i*)(v4i*)ldst, 0, 0);
}
#else
#define USE_ASYNC_LDS 0
#endif

__device__ __forceinline__ unsigned int pack_f16(float a, float b) {
  _Float16 ha = (_Float16)a;
  _Float16 hb = (_Float16)b;
  unsigned int ua = (unsigned int)__builtin_bit_cast(unsigned short, ha);
  unsigned int ub = (unsigned int)__builtin_bit_cast(unsigned short, hb);
  return ua | (ub << 16);
}

// ---------------------------------------------------------------------------
// Kernel 1: per-channel batch stats -> fold BN into scale/bias
// ---------------------------------------------------------------------------
__global__ __launch_bounds__(256) void bn_stats_kernel(
    const float* __restrict__ x, const float* __restrict__ gamma,
    const float* __restrict__ beta, float* __restrict__ scale,
    float* __restrict__ bias) {
  const int c = blockIdx.x;
  const int t = threadIdx.x;
  float s = 0.f, q = 0.f;
  for (int b = 0; b < B_; ++b) {
    const float* p = x + ((size_t)b * CIN + c) * HW;
    for (int i = t; i < HW; i += 256) {
      float v = p[i];
      s += v;
      q += v * v;
    }
  }
  __shared__ float rs[256];
  __shared__ float rq[256];
  rs[t] = s; rq[t] = q;
  __syncthreads();
  for (int st = 128; st > 0; st >>= 1) {
    if (t < st) { rs[t] += rs[t + st]; rq[t] += rq[t + st]; }
    __syncthreads();
  }
  if (t == 0) {
    const float inv_n = 1.0f / (float)(B_ * HW);
    float mean = rs[0] * inv_n;
    float var  = rq[0] * inv_n - mean * mean;   // biased variance
    float sc   = gamma[c] * rsqrtf(var + EPSF);
    scale[c] = sc;
    bias[c]  = beta[c] - mean * sc;
  }
}

// ---------------------------------------------------------------------------
// Kernel 2: quantize weights, dequant, store f16 K-major [Cin][Cout]
// (int grid in [-128,127] * scale is exactly representable in f16)
// ---------------------------------------------------------------------------
__global__ __launch_bounds__(256) void quant_w_kernel(
    const float* __restrict__ w, const float* __restrict__ op_scale,
    const float* __restrict__ op_scale_bw, unsigned short* __restrict__ wq) {
  const int idx = blockIdx.x * 256 + threadIdx.x;  // idx = c*COUT + o
  const int c = idx >> 8;
  const int o = idx & 255;
  const float inv_s = 1.0f / op_scale[0];
  float v = rintf(w[o * CIN + c] * inv_s);         // round-half-even like jnp.round
  v = fminf(fmaxf(v, -128.f), 127.f) * op_scale_bw[0];
  _Float16 hv = (_Float16)v;
  wq[idx] = __builtin_bit_cast(unsigned short, hv);
}

// ---------------------------------------------------------------------------
// Kernel 3: fused BN-apply + 1x1 conv (f16 WMMA, f32 accum) + 2x2 avg pool
// Block: 256 thr (8 waves) covers 64 pixels (16 pool quads) x 256 Cout.
// Double-buffered LDS; weight tiles streamed with async global->LDS copies.
// ---------------------------------------------------------------------------
__global__ __launch_bounds__(256) void fused_gemm_pool_kernel(
    const float* __restrict__ x, const float* __restrict__ scale,
    const float* __restrict__ bias, unsigned short* __restrict__ wq,
    float* __restrict__ out) {
  const int t    = threadIdx.x;
  const int bimg = blockIdx.x / 49;
  const int tile = blockIdx.x % 49;   // 16 pool quads per tile (49*16 = 784)

  __shared__ float sS[CIN];
  __shared__ float sB[CIN];
  __shared__ __align__(16) unsigned int sBuf[2 * HSZ + 2 * WSZ];
  unsigned int* hbase = sBuf;            // + cur*HSZ
  unsigned int* wbase = sBuf + 2 * HSZ;  // + cur*WSZ

  for (int i = t; i < CIN; i += 256) { sS[i] = scale[i]; sB[i] = bias[i]; }

  // activation staging: thread loads 8 channels of one pixel (quad-major order)
  const int p   = t & 63;
  const int cb  = (t >> 6) << 3;                 // {0,8,16,24}
  const int qg  = tile * 16 + (p >> 2);          // global pool-quad
  const int sub = p & 3;
  const int ih  = 2 * (qg / PH) + (sub >> 1);
  const int iw  = 2 * (qg % PH) + (sub & 1);
  const size_t xoff = ((size_t)bimg * CIN + cb) * HW + (size_t)ih * WW + iw;

  // weight staging: thread copies 64 bytes of one K-row
  const int wr  = t >> 3;         // K row 0..31
  const int wcb = (t & 7) << 5;   // col base (f16): 0,32,...,224

  // wave tiling: 2 M-groups (32 px) x 4 N-groups (64 oc)
  const int lane = t & 31;
  const int wv   = t >> 5;
  const int m0   = (wv & 1) << 5;
  const int n0   = (wv >> 1) << 6;

  const v8f vzero = {0.f, 0.f, 0.f, 0.f, 0.f, 0.f, 0.f, 0.f};
  v8f acc[2][4];
#pragma unroll
  for (int i = 0; i < 2; ++i)
#pragma unroll
    for (int j = 0; j < 4; ++j) acc[i][j] = vzero;

  // ---- prologue: start copy of weight tile 0 ----
  {
    const unsigned short* gsrc = wq + (size_t)wr * COUT + wcb;
    unsigned int* ldst = wbase + wr * 132 + (wcb >> 1);
#if USE_ASYNC_LDS
#pragma unroll
    for (int jj = 0; jj < 4; ++jj) async_b128(gsrc + jj * 8, ldst + jj * 4);
#else
    const uint4* s4 = (const uint4*)gsrc;
    uint4* d4 = (uint4*)ldst;
#pragma unroll
    for (int jj = 0; jj < 4; ++jj) d4[jj] = s4[jj];
#endif
  }

  for (int ks = 0; ks < KSTEPS; ++ks) {
    const int k0  = ks << 5;
    const int cur = ks & 1;
    // ---- stage normalized activations (BN + ReLU + f16) into h-buffer ----
    {
      const float* xp = x + xoff + (size_t)k0 * HW;
      unsigned int* hA = hbase + cur * HSZ;
#pragma unroll
      for (int jj = 0; jj < 4; ++jj) {
        float f0 = xp[(size_t)(2 * jj) * HW];
        float f1 = xp[(size_t)(2 * jj + 1) * HW];
        const int c0 = k0 + cb + 2 * jj;
        f0 = fmaxf(fmaf(f0, sS[c0],     sB[c0]),     0.f);
        f1 = fmaxf(fmaf(f1, sS[c0 + 1], sB[c0 + 1]), 0.f);
        hA[p * 20 + (cb >> 1) + jj] = pack_f16(f0, f1);
      }
    }
    if (k0 + 32 < CIN)  // global_prefetch_b8 next activation slab
      __builtin_prefetch(x + xoff + (size_t)(k0 + 32) * HW, 0, 1);

#if USE_ASYNC_LDS
    __builtin_amdgcn_s_wait_asynccnt(0);  // weight tile ks resident in w-buffer
#endif
    __syncthreads();

    // ---- issue copy of weight tile ks+1 (safe: all waves passed barrier) ----
    if (ks + 1 < KSTEPS) {
      const unsigned short* gsrc = wq + (size_t)(k0 + 32 + wr) * COUT + wcb;
      unsigned int* ldst = wbase + (cur ^ 1) * WSZ + wr * 132 + (wcb >> 1);
#if USE_ASYNC_LDS
#pragma unroll
      for (int jj = 0; jj < 4; ++jj) async_b128(gsrc + jj * 8, ldst + jj * 4);
#else
      const uint4* s4 = (const uint4*)gsrc;
      uint4* d4 = (uint4*)ldst;
#pragma unroll
      for (int jj = 0; jj < 4; ++jj) d4[jj] = s4[jj];
#endif
    }

    // ---- fragments ----
    const unsigned int* hA = hbase + cur * HSZ;
    const unsigned int* wB = wbase + cur * WSZ;
    Frag32B a[2];
    const int hs = (lane >> 4) << 2;  // upper half-wave: K 8..15 region
#pragma unroll
    for (int mi = 0; mi < 2; ++mi) {
      const unsigned int* rowp = hA + (m0 + mi * 16 + (lane & 15)) * 20 + hs;
#pragma unroll
      for (int u = 0; u < 4; ++u) {
        a[mi].u[u]     = rowp[u];       // K 0..7  / 8..15
        a[mi].u[4 + u] = rowp[8 + u];   // K 16..23 / 24..31
      }
    }
    Frag32B bf[4];
#pragma unroll
    for (int ni = 0; ni < 4; ++ni) {
      const unsigned short* bp = (const unsigned short*)wB + n0 + ni * 16 + (lane & 15);
#pragma unroll
      for (int u = 0; u < 8; ++u) {
        const int kk = ((u >> 2) << 4) + ((lane >> 4) << 3) + ((u & 3) << 1);
        bf[ni].u[u] = (unsigned int)bp[kk * 264] | ((unsigned int)bp[(kk + 1) * 264] << 16);
      }
    }
    // ---- 8 WMMAs ----
#pragma unroll
    for (int ni = 0; ni < 4; ++ni)
#pragma unroll
      for (int mi = 0; mi < 2; ++mi)
        acc[mi][ni] = __builtin_amdgcn_wmma_f32_16x16x32_f16(
            false, a[mi].v, false, bf[ni].v, (short)0, acc[mi][ni], false, false);
  }

  // ---- 2x2 avg pool: quad = 4 consecutive M rows = 4 accum components ----
  __syncthreads();
  float* pool = (float*)sBuf;  // reuse LDS: [256 oc][17] (padded)
#pragma unroll
  for (int mi = 0; mi < 2; ++mi) {
#pragma unroll
    for (int ni = 0; ni < 4; ++ni) {
      const v8f c = acc[mi][ni];
      const float s0 = (c[0] + c[1] + c[2] + c[3]) * 0.25f;
      const float s1 = (c[4] + c[5] + c[6] + c[7]) * 0.25f;
      const int oc = n0 + ni * 16 + (lane & 15);
      const int q  = ((m0 + mi * 16) >> 2) + ((lane >> 4) << 1);
      pool[oc * 17 + q]     = s0;
      pool[oc * 17 + q + 1] = s1;
    }
  }
  __syncthreads();

  // ---- coalesced transposed store ----
  const int qs = t & 15;
  const int qglob = tile * 16 + qs;
#pragma unroll
  for (int ocb = 0; ocb < 16; ++ocb) {
    const int oc = (t >> 4) + (ocb << 4);
    out[((size_t)bimg * COUT + oc) * PQ + qglob] = pool[oc * 17 + qs];
  }
}

// ---------------------------------------------------------------------------
extern "C" void kernel_launch(void* const* d_in, const int* in_sizes, int n_in,
                              void* d_out, int out_size, void* d_ws, size_t ws_size,
                              hipStream_t stream) {
  (void)in_sizes; (void)n_in; (void)out_size; (void)ws_size;
  const float* x           = (const float*)d_in[0];
  const float* gamma       = (const float*)d_in[1];
  const float* beta        = (const float*)d_in[2];
  const float* w           = (const float*)d_in[3];
  const float* op_scale    = (const float*)d_in[4];
  const float* op_scale_bw = (const float*)d_in[5];
  float* out = (float*)d_out;

  float* scale = (float*)d_ws;                                   // 512 f32
  float* bias  = scale + CIN;                                    // 512 f32
  unsigned short* wq = (unsigned short*)((char*)d_ws + 4096);    // 512x256 f16, K-major

  bn_stats_kernel<<<CIN, 256, 0, stream>>>(x, gamma, beta, scale, bias);
  quant_w_kernel<<<(CIN * COUT) / 256, 256, 0, stream>>>(w, op_scale, op_scale_bw, wq);
  fused_gemm_pool_kernel<<<B_ * 49, 256, 0, stream>>>(x, scale, bias, wq, out);
}
